// StridedAttention_19662360281650
// MI455X (gfx1250) — matrix operand order repaired and checked by reading, not compiled, not run on hardware
//
#include <hip/hip_runtime.h>

// ---------------------------------------------------------------------------
// Strided multi-head attention for MI455X (gfx1250, wave32, WMMA bf16)
//   B=2, S=8192, D=512, H=8, HD=64, STRIDE=4 -> Sk=2048
// ---------------------------------------------------------------------------

typedef __attribute__((ext_vector_type(16))) __bf16 v16bf;
typedef __attribute__((ext_vector_type(8)))  float  v8f;
typedef __attribute__((ext_vector_type(2)))  int    v2i_t;

union FragAB { v16bf v; unsigned int u[8]; };

// hardware RNE convert (v_cvt bf16) instead of manual integer rounding
__device__ __forceinline__ unsigned short f2bf(float f) {
  return __builtin_bit_cast(unsigned short, (__bf16)f);
}
__device__ __forceinline__ unsigned int pack2bf(float lo, float hi) {
  return (unsigned int)f2bf(lo) | ((unsigned int)f2bf(hi) << 16);
}

#define S_LEN   8192
#define D_MODEL 512
#define N_HEADS 8
#define HEAD_D  64
#define SK_LEN  2048

// ---------------------------------------------------------------------------
// Async global->LDS (ASYNCcnt path). Builtin confirmed present; signature is
// (v2i addrspace(1)*, v2i addrspace(3)*, imm offset, imm cpol).
// ---------------------------------------------------------------------------
#if defined(__has_builtin)
#  if __has_builtin(__builtin_amdgcn_global_load_async_to_lds_b64)
#    define ASYNC_LDS 1
#  endif
#endif
#ifndef ASYNC_LDS
#  define ASYNC_LDS 0
#endif

typedef __attribute__((address_space(1))) v2i_t* g64p;
typedef __attribute__((address_space(3))) v2i_t* l64p;

__device__ __forceinline__ void cp_g2l_b64(void* lds, const void* g) {
#if ASYNC_LDS
  __builtin_amdgcn_global_load_async_to_lds_b64((g64p)g, (l64p)lds, 0, 0);
#else
  *(unsigned long long*)lds = *(const unsigned long long*)g;
#endif
}
__device__ __forceinline__ void cp_wait_async() {
#if ASYNC_LDS
#  if __has_builtin(__builtin_amdgcn_s_wait_asynccnt)
  __builtin_amdgcn_s_wait_asynccnt(0);
#  else
  asm volatile("s_wait_asynccnt 0" ::: "memory");
#  endif
#endif
}

// ---------------------------------------------------------------------------
// Kernel 1: projection  out = act @ W^T + b -> bf16 head-major [B,H,Sout,64].
// K/V project only the strided source rows (4x fewer FLOPs).
// Block 256 thr (8 waves), tile 128x64, K-step 32.
// Wave w: rows (w>>1)*32 (two 16-row tiles), cols (w&1)*32 (two 16-col tiles)
//   -> 4 WMMA per K-step from 2 A-frags x 2 B-frags.
// SOUT/STRIDE are compile-time: epilogue div/mod become shifts/masks.
// ---------------------------------------------------------------------------
template<int SOUT, int STRIDE>
__global__ __launch_bounds__(256) void qkv_gemm(
    const float* __restrict__ A,      // [B*S, 512]
    const float* __restrict__ W,      // [512, 512]
    const float* __restrict__ bias,   // [512]
    unsigned short* __restrict__ out) // bf16 [B, H, SOUT, 64]
{
  const int tileM = blockIdx.x;
  const int tileN = blockIdx.y;
  const int tid   = threadIdx.x;
  const int wave  = tid >> 5;
  const int lane  = tid & 31;
  const int half  = lane >> 4;
  const int l16   = lane & 15;

  __shared__ unsigned short As[128][34];   // odd uint stride -> no bank conflicts
  __shared__ unsigned short Ws[64][34];

  const int rt = wave >> 1;              // row group 0..3 (32 rows each)
  const int cg = wave & 1;               // col group 0..1 (32 cols each)

  v8f acc[2][2] = {};

  // A loader: thread -> one row, 16 consecutive k
  const int ldRowA = tid >> 1;           // 0..127
  const int ldKA   = (tid & 1) * 16;
  // W loader: thread -> one row, 8 consecutive k
  const int ldRowW = tid >> 2;           // 0..63
  const int ldKW   = (tid & 3) * 8;

  const int mOut = tileM * 128 + ldRowA;
  const int bb   = mOut / SOUT;          // compile-time pow2 -> shift
  const int sOut = mOut % SOUT;
  const float* Arow = A + (size_t)(bb * S_LEN + sOut * STRIDE) * D_MODEL;
  const float* Wrow = W + (size_t)(tileN * 64 + ldRowW) * D_MODEL;

  for (int k0 = 0; k0 < D_MODEL; k0 += 32) {
    __builtin_prefetch(Arow + k0 + 64, 0, 0);   // global_prefetch_b8
    #pragma unroll
    for (int i = 0; i < 16; i += 2)
      *(unsigned int*)&As[ldRowA][ldKA + i] =
          pack2bf(Arow[k0 + ldKA + i], Arow[k0 + ldKA + i + 1]);
    #pragma unroll
    for (int i = 0; i < 8; i += 2)
      *(unsigned int*)&Ws[ldRowW][ldKW + i] =
          pack2bf(Wrow[k0 + ldKW + i], Wrow[k0 + ldKW + i + 1]);
    __syncthreads();

    FragAB fa[2], fb[2];
    #pragma unroll
    for (int s = 0; s < 2; ++s)
      #pragma unroll
      for (int v = 0; v < 8; ++v) {
        const int kkA = 2 * (v & 3) + 16 * (v >> 2) + 8 * half;
        fa[s].u[v] = *(const unsigned int*)&As[rt * 32 + s * 16 + l16][kkA];
        const int kkB = 2 * (v & 3) + 16 * (v >> 2) + 8 * half;
        fb[s].u[v] = *(const unsigned int*)&Ws[cg * 32 + s * 16 + l16][kkB];
      }
    #pragma unroll
    for (int rs = 0; rs < 2; ++rs)
      #pragma unroll
      for (int cs = 0; cs < 2; ++cs)
        acc[rs][cs] = __builtin_amdgcn_wmma_f32_16x16x32_bf16(
            false, fa[rs].v, false, fb[cs].v, (short)0, acc[rs][cs], false, false);
    __syncthreads();
  }

  #pragma unroll
  for (int rs = 0; rs < 2; ++rs)
    #pragma unroll
    for (int cs = 0; cs < 2; ++cs) {
      const int n = tileN * 64 + cg * 32 + cs * 16 + l16;
      const int h = n >> 6, d = n & 63;
      const float bval = bias[n];
      #pragma unroll
      for (int r = 0; r < 8; ++r) {
        const int m  = tileM * 128 + rt * 32 + rs * 16 + r + 8 * half;
        const int b2 = m / SOUT;
        const int s2 = m % SOUT;
        out[(((size_t)(b2 * N_HEADS + h)) * SOUT + s2) * HEAD_D + d] =
            f2bf(acc[rs][cs][r] + bval);
      }
    }
}

// ---------------------------------------------------------------------------
// Kernel 2: flash attention per (b,h, 64-query tile). 4 waves x 16 q-rows.
// Chunk = 64 keys (async-staged): scores 8 WMMA + P*V 8 WMMA per chunk.
// Online softmax fp32 via intra-half shfl reductions. Output bf16 [B,S,512].
// ---------------------------------------------------------------------------
__global__ __launch_bounds__(128) void attn_kernel(
    const unsigned short* __restrict__ Qb, // [B,H,S,64]
    const unsigned short* __restrict__ Kb, // [B,H,Sk,64]
    const unsigned short* __restrict__ Vb, // [B,H,Sk,64]
    unsigned short* __restrict__ O)        // [B,S,512]
{
  const int bh   = blockIdx.x >> 7;       // S/64 = 128 q-tiles per head
  const int qt   = blockIdx.x & 127;
  const int b    = bh >> 3, h = bh & 7;
  const int tid  = threadIdx.x;
  const int wave = tid >> 5, lane = tid & 31;
  const int half = lane >> 4, l16 = lane & 15;

  // stride 68 ushorts = 136B: rows 8B-aligned (async b64 ok), 34-uint stride
  // -> conflict-free 32-bit fragment reads.
  __shared__ unsigned short Kch[64][68];     // [key][d]
  __shared__ unsigned short Vch[64][68];     // [key][d]
  __shared__ unsigned short Ps[4][16][66];   // per-wave P scratch (C->A relayout)

  const unsigned short* Qbase =
      Qb + (((size_t)bh) * S_LEN + qt * 64 + wave * 16) * HEAD_D;
  FragAB fq[2];
  #pragma unroll
  for (int ds = 0; ds < 2; ++ds)
    #pragma unroll
    for (int v = 0; v < 8; ++v) {
      const int dd = ds * 32 + 2 * (v & 3) + 16 * (v >> 2) + 8 * half;
      fq[ds].u[v] = *(const unsigned int*)&Qbase[(size_t)l16 * HEAD_D + dd];
    }

  float mrow[8], lrow[8];
  #pragma unroll
  for (int r = 0; r < 8; ++r) { mrow[r] = -1e30f; lrow[r] = 0.f; }
  v8f acc[4] = {};

  const unsigned short* Kg = Kb + ((size_t)bh) * SK_LEN * HEAD_D;
  const unsigned short* Vg = Vb + ((size_t)bh) * SK_LEN * HEAD_D;

  for (int kc = 0; kc < SK_LEN; kc += 64) {
    // async stage: 64 keys x 128B = 1024 x 8B segments each for K and V
    #pragma unroll
    for (int i = 0; i < 8; ++i) {
      const int idx = tid + i * 128;
      const int key = idx >> 4, seg = idx & 15;
      cp_g2l_b64(&Kch[key][seg * 4], Kg + (size_t)(kc + key) * HEAD_D + seg * 4);
      cp_g2l_b64(&Vch[key][seg * 4], Vg + (size_t)(kc + key) * HEAD_D + seg * 4);
    }
    cp_wait_async();
    __syncthreads();

    // scores: four 16-key tiles, head dim reduced in 2 WMMA each
    v8f sc[4] = {};
    #pragma unroll
    for (int t = 0; t < 4; ++t)
      #pragma unroll
      for (int ds = 0; ds < 2; ++ds) {
        FragAB fk;
        #pragma unroll
        for (int v = 0; v < 8; ++v) {
          const int dd = ds * 32 + 2 * (v & 3) + 16 * (v >> 2) + 8 * half;
          fk.u[v] = *(const unsigned int*)&Kch[t * 16 + l16][dd];
        }
        sc[t] = __builtin_amdgcn_wmma_f32_16x16x32_bf16(
            false, fq[ds].v, false, fk.v, (short)0, sc[t], false, false);
      }

    // online softmax over the 64 new columns (scale = 1/sqrt(64) = 0.125)
    float mnew[8], rescale[8];
    #pragma unroll
    for (int r = 0; r < 8; ++r) {
      float x = fmaxf(fmaxf(sc[0][r], sc[1][r]), fmaxf(sc[2][r], sc[3][r])) * 0.125f;
      x = fmaxf(x, __shfl_xor(x, 1, 32));
      x = fmaxf(x, __shfl_xor(x, 2, 32));
      x = fmaxf(x, __shfl_xor(x, 4, 32));
      x = fmaxf(x, __shfl_xor(x, 8, 32));   // masks<16 stay within lane halves
      mnew[r]    = fmaxf(mrow[r], x);
      rescale[r] = __expf(mrow[r] - mnew[r]);
      mrow[r]    = mnew[r];
    }
    #pragma unroll
    for (int r = 0; r < 8; ++r) {
      float p[4], rs = 0.f;
      #pragma unroll
      for (int t = 0; t < 4; ++t) {
        p[t] = __expf(sc[t][r] * 0.125f - mnew[r]);
        rs += p[t];
        Ps[wave][r + 8 * half][t * 16 + l16] = f2bf(p[t]);
      }
      rs += __shfl_xor(rs, 1, 32);
      rs += __shfl_xor(rs, 2, 32);
      rs += __shfl_xor(rs, 4, 32);
      rs += __shfl_xor(rs, 8, 32);
      lrow[r] = lrow[r] * rescale[r] + rs;
      #pragma unroll
      for (int j = 0; j < 4; ++j) acc[j][r] *= rescale[r];
    }

    // P (16x64) x V (64x64): two K=32 halves (same-wave DScnt ordering)
    #pragma unroll
    for (int kh = 0; kh < 2; ++kh) {
      FragAB fp;
      #pragma unroll
      for (int v = 0; v < 8; ++v) {
        const int kk = kh * 32 + 2 * (v & 3) + 16 * (v >> 2) + 8 * half;
        fp.u[v] = *(const unsigned int*)&Ps[wave][l16][kk];
      }
      #pragma unroll
      for (int j = 0; j < 4; ++j) {
        FragAB fv;
        #pragma unroll
        for (int v = 0; v < 8; ++v) {
          const int k1   = kh * 32 + 2 * (v & 3) + 16 * (v >> 2) + 8 * half;   // key index
          const int dcol = j * 16 + l16;
          fv.u[v] = (unsigned int)Vch[k1][dcol] |
                    ((unsigned int)Vch[k1 + 1][dcol] << 16);
        }
        acc[j] = __builtin_amdgcn_wmma_f32_16x16x32_bf16(
            false, fp.v, false, fv.v, (short)0, acc[j], false, false);
      }
    }
    __syncthreads();
  }

  #pragma unroll
  for (int r = 0; r < 8; ++r) {
    const float inv = 1.0f / lrow[r];
    const int s = qt * 64 + wave * 16 + r + 8 * half;
    const size_t obase = ((size_t)b * S_LEN + s) * D_MODEL + h * HEAD_D;
    #pragma unroll
    for (int j = 0; j < 4; ++j)
      O[obase + j * 16 + l16] = f2bf(acc[j][r] * inv);
  }
}

// ---------------------------------------------------------------------------
// Kernel 3: output projection  d_out = attn(bf16) @ Wo^T + bo  (fp32 out)
// Same 128x64 tiling; bf16 A tile staged with async b64 copies.
// ---------------------------------------------------------------------------
__global__ __launch_bounds__(256) void out_gemm(
    const unsigned short* __restrict__ A, // bf16 [B*S, 512]
    const float* __restrict__ W,          // [512, 512]
    const float* __restrict__ bias,       // [512]
    float* __restrict__ out)              // [B*S, 512]
{
  const int tileM = blockIdx.x;
  const int tileN = blockIdx.y;
  const int tid   = threadIdx.x;
  const int wave  = tid >> 5;
  const int lane  = tid & 31;
  const int half  = lane >> 4;
  const int l16   = lane & 15;

  // stride 36 ushorts = 72B: rows 8B-aligned for async b64; 18-uint stride
  // -> conflict-free fragment reads.
  __shared__ unsigned short As[128][36];
  __shared__ unsigned short Ws[64][34];

  const int rt = wave >> 1;
  const int cg = wave & 1;

  v8f acc[2][2] = {};

  const int ldRowW = tid >> 2;
  const int ldKW   = (tid & 3) * 8;
  const float* Wrow = W + (size_t)(tileN * 64 + ldRowW) * D_MODEL;

  for (int k0 = 0; k0 < D_MODEL; k0 += 32) {
    // A tile: 128 rows x 64B = 1024 x 8B segments, 4 per thread
    #pragma unroll
    for (int i = 0; i < 4; ++i) {
      const int idx = tid + i * 256;
      const int row = idx >> 3, seg = idx & 7;
      cp_g2l_b64(&As[row][seg * 4],
                 A + (size_t)(tileM * 128 + row) * D_MODEL + k0 + seg * 4);
    }
    #pragma unroll
    for (int i = 0; i < 8; i += 2)
      *(unsigned int*)&Ws[ldRowW][ldKW + i] =
          pack2bf(Wrow[k0 + ldKW + i], Wrow[k0 + ldKW + i + 1]);
    cp_wait_async();
    __syncthreads();

    FragAB fa[2], fb[2];
    #pragma unroll
    for (int s = 0; s < 2; ++s)
      #pragma unroll
      for (int v = 0; v < 8; ++v) {
        const int kkA = 2 * (v & 3) + 16 * (v >> 2) + 8 * half;
        fa[s].u[v] = *(const unsigned int*)&As[rt * 32 + s * 16 + l16][kkA];
        const int kkB = 2 * (v & 3) + 16 * (v >> 2) + 8 * half;
        fb[s].u[v] = *(const unsigned int*)&Ws[cg * 32 + s * 16 + l16][kkB];
      }
    #pragma unroll
    for (int rs = 0; rs < 2; ++rs)
      #pragma unroll
      for (int cs = 0; cs < 2; ++cs)
        acc[rs][cs] = __builtin_amdgcn_wmma_f32_16x16x32_bf16(
            false, fa[rs].v, false, fb[cs].v, (short)0, acc[rs][cs], false, false);
    __syncthreads();
  }

  #pragma unroll
  for (int rs = 0; rs < 2; ++rs)
    #pragma unroll
    for (int cs = 0; cs < 2; ++cs) {
      const int n = tileN * 64 + cg * 32 + cs * 16 + l16;
      const float bval = bias[n];
      #pragma unroll
      for (int r = 0; r < 8; ++r) {
        const int m = tileM * 128 + rt * 32 + rs * 16 + r + 8 * half;
        out[(size_t)m * D_MODEL + n] = acc[rs][cs][r] + bval;
      }
    }
}

// ---------------------------------------------------------------------------
extern "C" void kernel_launch(void* const* d_in, const int* in_sizes, int n_in,
                              void* d_out, int out_size, void* d_ws, size_t ws_size,
                              hipStream_t stream) {
  const float* query = (const float*)d_in[0];
  const float* key_  = (const float*)d_in[1];
  const float* value = (const float*)d_in[2];
  const float* Wq = (const float*)d_in[3];
  const float* bq = (const float*)d_in[4];
  const float* Wk = (const float*)d_in[5];
  const float* bk = (const float*)d_in[6];
  const float* Wv = (const float*)d_in[7];
  const float* bv = (const float*)d_in[8];
  const float* Wo = (const float*)d_in[9];
  const float* bo = (const float*)d_in[10];

  // workspace layout (bf16 = ushort bits)
  unsigned short* Qb = (unsigned short*)d_ws;                      // 2*8*8192*64
  unsigned short* Kb = Qb + (size_t)2 * N_HEADS * S_LEN * HEAD_D;  // 2*8*2048*64
  unsigned short* Vb = Kb + (size_t)2 * N_HEADS * SK_LEN * HEAD_D;
  unsigned short* Ao = Vb + (size_t)2 * N_HEADS * SK_LEN * HEAD_D; // 2*8192*512

  qkv_gemm<S_LEN, 1><<<dim3(128, 8), dim3(256), 0, stream>>>(query, Wq, bq, Qb);
  qkv_gemm<SK_LEN, 4><<<dim3(32, 8), dim3(256), 0, stream>>>(key_, Wk, bk, Kb);
  qkv_gemm<SK_LEN, 4><<<dim3(32, 8), dim3(256), 0, stream>>>(value, Wv, bv, Vb);

  attn_kernel<<<dim3(2 * N_HEADS * (S_LEN / 64)), dim3(128), 0, stream>>>(Qb, Kb, Vb, Ao);

  out_gemm<<<dim3(128, 8), dim3(256), 0, stream>>>(Ao, Wo, bo, (float*)d_out);
}
